// MixtralMoE_51625506898147
// MI455X (gfx1250) — matrix operand order repaired and checked by reading, not compiled, not run on hardware
//
#include <hip/hip_runtime.h>
#include <hip/hip_bf16.h>

// ---------------- problem constants ----------------
#define E_     8
#define TOPK_  2
#define H_     1024
#define F_     3584
#define T_     2048
#define SLOTS  (T_ * TOPK_)      // 4096
#define PSLOTS 4608              // slot space padded: per-expert 32-aligned bases
#define PR16   (PSLOTS / 16)     // 288 16-row fragment blocks

#define KT_H   (H_ / 32)         // 32  k-tiles over H
#define KT_F   (F_ / 32)         // 112 k-tiles over F
#define N16_F  (F_ / 16)         // 224 16-col tiles over F (w1/w3)
#define N16_H  (H_ / 16)         // 64  16-col tiles over H (w2)

// GEMM tiling: 64 rows x 128 cols per block; 8 waves, each 2 M-frags x 2 N16-tiles
#define BM 64
#define BN 128

typedef __attribute__((ext_vector_type(16))) __bf16 v16bf;
typedef __attribute__((ext_vector_type(8)))  float  v8f;

static __device__ inline v8f wmma_bf16(v16bf a, v16bf b, v8f c) {
  return __builtin_amdgcn_wmma_f32_16x16x32_bf16(false, a, false, b, (short)0, c,
                                                 false, false);
}

static __device__ inline void store_hilo(__bf16* __restrict__ hp, __bf16* __restrict__ lp,
                                         size_t idx, float v) {
  __bf16 h = (__bf16)v;
  hp[idx] = h;
  lp[idx] = (__bf16)(v - (float)h);
}

// ---------------- workspace layout (bytes) ----------------
static constexpr size_t OFF_COUNTS  = 0;
static constexpr size_t OFF_CURSOR  = 64;
static constexpr size_t OFF_OFFSETS = 128;
static constexpr size_t OFF_TOPIDX  = 256;
static constexpr size_t OFF_TOPW    = OFF_TOPIDX + 4ull * 2 * T_;
static constexpr size_t OFF_TOKEN   = OFF_TOPW + 4ull * 2 * T_;
static constexpr size_t OFF_WOF     = OFF_TOKEN + 4ull * PSLOTS;
static constexpr size_t OFF_AXH     = (OFF_WOF + 4ull * PSLOTS + 255) & ~255ull;

static constexpr size_t AX_ELEMS  = (size_t)PR16 * KT_H * 512;        //  4.7M
static constexpr size_t ACT_ELEMS = (size_t)PR16 * KT_F * 512;        // 16.5M
static constexpr size_t W13_ELEMS = (size_t)E_ * N16_F * KT_H * 512;  // 29.4M
static constexpr size_t W2_ELEMS  = (size_t)E_ * N16_H * KT_F * 512;  // 29.4M

static constexpr size_t OFF_AXL  = OFF_AXH  + 2 * AX_ELEMS;
static constexpr size_t OFF_ACTH = OFF_AXL  + 2 * AX_ELEMS;
static constexpr size_t OFF_ACTL = OFF_ACTH + 2 * ACT_ELEMS;
static constexpr size_t OFF_W1H  = OFF_ACTL + 2 * ACT_ELEMS;
static constexpr size_t OFF_W1L  = OFF_W1H + 2 * W13_ELEMS;
static constexpr size_t OFF_W3H  = OFF_W1L + 2 * W13_ELEMS;
static constexpr size_t OFF_W3L  = OFF_W3H + 2 * W13_ELEMS;
static constexpr size_t OFF_W2H  = OFF_W3L + 2 * W13_ELEMS;
static constexpr size_t OFF_W2L  = OFF_W2H + 2 * W2_ELEMS;

// ---------------- kernel 0: zero out / counts / token pad ----------------
__global__ void moe_zero_kernel(float* __restrict__ out, int* __restrict__ counts,
                                int* __restrict__ tokenOf) {
  int i = blockIdx.x * blockDim.x + threadIdx.x;
  if (i < T_ * H_) out[i] = 0.f;
  if (i < PSLOTS) tokenOf[i] = -1;
  if (i < E_) counts[i] = 0;
}

// ---------------- kernel 1: router (one wave32 per token) ----------------
__global__ __launch_bounds__(256) void moe_router_kernel(
    const float* __restrict__ x, const float* __restrict__ gw,
    int* __restrict__ counts, int* __restrict__ topIdx, float* __restrict__ topW) {
  int wave = threadIdx.x >> 5;
  int lane = threadIdx.x & 31;
  int t = blockIdx.x * 8 + wave;
  if (t >= T_) return;
  const float* xr = x + (size_t)t * H_;

  float acc[E_];
#pragma unroll
  for (int e = 0; e < E_; ++e) acc[e] = 0.f;
  for (int j = lane; j < H_; j += 32) {
    float xv = xr[j];
#pragma unroll
    for (int e = 0; e < E_; ++e) acc[e] += xv * gw[e * H_ + j];
  }
#pragma unroll
  for (int e = 0; e < E_; ++e) {
#pragma unroll
    for (int m = 16; m >= 1; m >>= 1) acc[e] += __shfl_xor(acc[e], m, 32);
  }
  if (lane == 0) {
    float mx = acc[0];
#pragma unroll
    for (int e = 1; e < E_; ++e) mx = fmaxf(mx, acc[e]);
    float p[E_];
#pragma unroll
    for (int e = 0; e < E_; ++e) p[e] = __expf(acc[e] - mx);
    int i1 = 0;
#pragma unroll
    for (int e = 1; e < E_; ++e) if (p[e] > p[i1]) i1 = e;
    int i2 = (i1 == 0) ? 1 : 0;
#pragma unroll
    for (int e = 0; e < E_; ++e) if (e != i1 && p[e] > p[i2]) i2 = e;
    float s = p[i1] + p[i2];
    topIdx[2 * t + 0] = i1;
    topIdx[2 * t + 1] = i2;
    topW[2 * t + 0] = p[i1] / s;
    topW[2 * t + 1] = p[i2] / s;
    atomicAdd(&counts[i1], 1);
    atomicAdd(&counts[i2], 1);
  }
}

// ---------------- kernel 2: scan with 32-aligned expert bases ----------------
__global__ void moe_scan_kernel(const int* __restrict__ counts, int* __restrict__ cursor,
                                int* __restrict__ offsets) {
  if (threadIdx.x == 0) {
    int o = 0;
#pragma unroll
    for (int e = 0; e < E_; ++e) {
      offsets[e] = o;
      o += counts[e];
      o = (o + 31) & ~31;  // 32-align: fragments never straddle experts
      cursor[e] = 0;
    }
    offsets[E_] = o;
  }
}

// ---------------- kernel 3: compact per-expert token lists ----------------
__global__ void moe_fill_kernel(const int* __restrict__ topIdx, const float* __restrict__ topW,
                                const int* __restrict__ offsets, int* __restrict__ cursor,
                                int* __restrict__ tokenOf, float* __restrict__ wOf) {
  int t = blockIdx.x * blockDim.x + threadIdx.x;
  if (t >= T_) return;
#pragma unroll
  for (int j = 0; j < TOPK_; ++j) {
    int e = topIdx[2 * t + j];
    int pos = atomicAdd(&cursor[e], 1);
    int slot = offsets[e] + pos;
    tokenOf[slot] = t;
    wOf[slot] = topW[2 * t + j];
  }
}

// ---------------- kernel 4: swizzle weights -> fragment-ready hi/lo bf16 planes ----
// B-fragment (KxN 32x16 bf16): lane = (n&15) + 16*((k>>3)&1); elem i = (k&7)+8*(k>>4).
__global__ __launch_bounds__(256) void moe_swizzle_w_kernel(
    const float* __restrict__ w, __bf16* __restrict__ hp, __bf16* __restrict__ lp,
    int N16, int KT, int K) {
  int l = blockIdx.x * 256 + threadIdx.x;  // [0, KT*512)
  int kt = l >> 9;
  int within = l & 511;
  int lane = within >> 4;
  int i = within & 15;
  int n = (blockIdx.y % N16) * 16 + (lane & 15);
  int e = blockIdx.y / N16;
  int k = kt * 32 + (i & 7) + 8 * (lane >> 4) + 16 * (i >> 3);
  float v = w[((size_t)e * (N16 * 16) + n) * K + k];
  size_t dst = ((size_t)blockIdx.y * KT + kt) * 512 + within;
  store_hilo(hp, lp, dst, v);
}

// ---------------- kernel 5: gather+swizzle routed X -> A fragment planes --------
// A-fragment (MxK 16x32 bf16): lane = m + 16*half, half=(k>>3)&1; i = (k&7)+8*(k>>4).
__global__ __launch_bounds__(256) void moe_swizzle_x_kernel(
    const float* __restrict__ x, const int* __restrict__ tokenOf,
    __bf16* __restrict__ hp, __bf16* __restrict__ lp) {
  int l = blockIdx.x * 256 + threadIdx.x;  // [0, KT_H*512)
  int kt = l >> 9;
  int within = l & 511;
  int lane = within >> 4;
  int i = within & 15;
  int r16 = blockIdx.y;
  int m = lane & 15;
  int half = lane >> 4;
  int k = (i & 7) | (half << 3) | ((i >> 3) << 4);
  int tok = tokenOf[r16 * 16 + m];
  float v = (tok >= 0) ? x[(size_t)tok * H_ + kt * 32 + k] : 0.f;
  size_t dst = ((size_t)r16 * KT_H + kt) * 512 + within;
  store_hilo(hp, lp, dst, v);
}

// ---------------- kernel 6: up-projection + SiLU gate (pure WMMA inner loop) ----
// wave = (wmg 0..1, wn 0..3); each wave: M-frags {wmg*2, wmg*2+1}, N16 {wn*2, wn*2+1}
__global__ __launch_bounds__(256) void moe_up_kernel(
    const __bf16* __restrict__ Axh, const __bf16* __restrict__ Axl,
    const __bf16* __restrict__ W1h, const __bf16* __restrict__ W1l,
    const __bf16* __restrict__ W3h, const __bf16* __restrict__ W3l,
    const int* __restrict__ counts, const int* __restrict__ offsets,
    __bf16* __restrict__ actH, __bf16* __restrict__ actL) {
  const int e = blockIdx.z;
  const int Ne = counts[e];
  const int m0 = blockIdx.y * BM;
  if (m0 >= Ne) return;  // uniform early-exit
  const int n0 = blockIdx.x * BN;
  const int base = offsets[e];

  const int wave = threadIdx.x >> 5;
  const int lane = threadIdx.x & 31;
  const int wmg = wave >> 2;  // 0..1 -> M-frag pair
  const int wn = wave & 3;    // 0..3 -> N16 pair
  const int half = lane >> 4;
  const int n = lane & 15;

  size_t aBase[2];
#pragma unroll
  for (int mi = 0; mi < 2; ++mi)
    aBase[mi] = ((size_t)(((base + m0) >> 4) + wmg * 2 + mi) * KT_H) * 512 + lane * 16;

  v8f cg[2][2] = {{{}, {}}, {{}, {}}};
  v8f cu[2][2] = {{{}, {}}, {{}, {}}};

  for (int kt = 0; kt < KT_H; ++kt) {
    v16bf a_h[2], a_l[2];
#pragma unroll
    for (int mi = 0; mi < 2; ++mi) {
      a_h[mi] = *(const v16bf*)(Axh + aBase[mi] + (size_t)kt * 512);
      a_l[mi] = *(const v16bf*)(Axl + aBase[mi] + (size_t)kt * 512);
    }
#pragma unroll
    for (int nt = 0; nt < 2; ++nt) {
      int n16 = (n0 >> 4) + wn * 2 + nt;
      size_t bIdx = (((size_t)e * N16_F + n16) * KT_H + kt) * 512 + lane * 16;
      {
        v16bf b_h = *(const v16bf*)(W1h + bIdx);
        v16bf b_l = *(const v16bf*)(W1l + bIdx);
        if (kt + 1 < KT_H) __builtin_prefetch(W1h + bIdx + 512, 0, 3);
#pragma unroll
        for (int mi = 0; mi < 2; ++mi) {
          cg[mi][nt] = wmma_bf16(a_h[mi], b_h, cg[mi][nt]);
          cg[mi][nt] = wmma_bf16(a_h[mi], b_l, cg[mi][nt]);
          cg[mi][nt] = wmma_bf16(a_l[mi], b_h, cg[mi][nt]);
        }
      }
      {
        v16bf b_h = *(const v16bf*)(W3h + bIdx);
        v16bf b_l = *(const v16bf*)(W3l + bIdx);
        if (kt + 1 < KT_H) __builtin_prefetch(W3h + bIdx + 512, 0, 3);
#pragma unroll
        for (int mi = 0; mi < 2; ++mi) {
          cu[mi][nt] = wmma_bf16(a_h[mi], b_h, cu[mi][nt]);
          cu[mi][nt] = wmma_bf16(a_h[mi], b_l, cu[mi][nt]);
          cu[mi][nt] = wmma_bf16(a_l[mi], b_h, cu[mi][nt]);
        }
      }
    }
  }

  // epilogue: act = silu(g)*u, stored directly in down-GEMM A-fragment layout
#pragma unroll
  for (int mi = 0; mi < 2; ++mi) {
#pragma unroll
    for (int nt = 0; nt < 2; ++nt) {
#pragma unroll
      for (int v = 0; v < 8; ++v) {
        int row = m0 + (wmg * 2 + mi) * 16 + v + 8 * half;  // C layout row
        if (row < Ne) {
          int absRow = base + row;
          int f = n0 + wn * 32 + nt * 16 + n;
          float gg = cg[mi][nt][v];
          float s = gg / (1.f + __expf(-gg));
          int c = f & 31;
          int dl = (absRow & 15) + 16 * ((c >> 3) & 1);
          int di = (c & 7) + 8 * ((c >> 4) & 1);
          size_t idx = (((size_t)(absRow >> 4) * KT_F + (f >> 5)) * 32 + dl) * 16 + di;
          store_hilo(actH, actL, idx, s * cu[mi][nt][v]);
        }
      }
    }
  }
}

// ---------------- kernel 7: down-projection + weighted scatter ----------------
__global__ __launch_bounds__(256) void moe_down_kernel(
    const __bf16* __restrict__ actH, const __bf16* __restrict__ actL,
    const __bf16* __restrict__ W2h, const __bf16* __restrict__ W2l,
    const int* __restrict__ counts, const int* __restrict__ offsets,
    const int* __restrict__ tokenOf, const float* __restrict__ wOf,
    float* __restrict__ out) {
  const int e = blockIdx.z;
  const int Ne = counts[e];
  const int m0 = blockIdx.y * BM;
  if (m0 >= Ne) return;
  const int n0 = blockIdx.x * BN;  // over H_
  const int base = offsets[e];

  const int wave = threadIdx.x >> 5;
  const int lane = threadIdx.x & 31;
  const int wmg = wave >> 2;
  const int wn = wave & 3;
  const int half = lane >> 4;
  const int n = lane & 15;

  size_t aBase[2];
#pragma unroll
  for (int mi = 0; mi < 2; ++mi)
    aBase[mi] = ((size_t)(((base + m0) >> 4) + wmg * 2 + mi) * KT_F) * 512 + lane * 16;

  v8f cy[2][2] = {{{}, {}}, {{}, {}}};

  for (int kt = 0; kt < KT_F; ++kt) {
    v16bf a_h[2], a_l[2];
#pragma unroll
    for (int mi = 0; mi < 2; ++mi) {
      a_h[mi] = *(const v16bf*)(actH + aBase[mi] + (size_t)kt * 512);
      a_l[mi] = *(const v16bf*)(actL + aBase[mi] + (size_t)kt * 512);
    }
#pragma unroll
    for (int nt = 0; nt < 2; ++nt) {
      int n16 = (n0 >> 4) + wn * 2 + nt;
      size_t bIdx = (((size_t)e * N16_H + n16) * KT_F + kt) * 512 + lane * 16;
      v16bf b_h = *(const v16bf*)(W2h + bIdx);
      v16bf b_l = *(const v16bf*)(W2l + bIdx);
      if (kt + 1 < KT_F) __builtin_prefetch(W2h + bIdx + 512, 0, 3);
#pragma unroll
      for (int mi = 0; mi < 2; ++mi) {
        cy[mi][nt] = wmma_bf16(a_h[mi], b_h, cy[mi][nt]);
        cy[mi][nt] = wmma_bf16(a_h[mi], b_l, cy[mi][nt]);
        cy[mi][nt] = wmma_bf16(a_l[mi], b_h, cy[mi][nt]);
      }
    }
  }

#pragma unroll
  for (int mi = 0; mi < 2; ++mi) {
#pragma unroll
    for (int nt = 0; nt < 2; ++nt) {
#pragma unroll
      for (int v = 0; v < 8; ++v) {
        int row = m0 + (wmg * 2 + mi) * 16 + v + 8 * half;
        if (row < Ne) {
          int tok = tokenOf[base + row];
          float w = wOf[base + row];
          int col = n0 + wn * 32 + nt * 16 + n;
          atomicAdd(&out[(size_t)tok * H_ + col], w * cy[mi][nt][v]);
        }
      }
    }
  }
}

// ---------------- host launcher ----------------
extern "C" void kernel_launch(void* const* d_in, const int* in_sizes, int n_in,
                              void* d_out, int out_size, void* d_ws, size_t ws_size,
                              hipStream_t stream) {
  const float* x  = (const float*)d_in[0];  // [1,2048,1024]
  const float* gw = (const float*)d_in[1];  // [8,1024]
  const float* w1 = (const float*)d_in[2];  // [8,3584,1024]
  const float* w3 = (const float*)d_in[3];  // [8,3584,1024]
  const float* w2 = (const float*)d_in[4];  // [8,1024,3584]
  float* out = (float*)d_out;               // [1,2048,1024]

  char* ws = (char*)d_ws;
  int*    counts  = (int*)(ws + OFF_COUNTS);
  int*    cursor  = (int*)(ws + OFF_CURSOR);
  int*    offsets = (int*)(ws + OFF_OFFSETS);
  int*    topIdx  = (int*)(ws + OFF_TOPIDX);
  float*  topW    = (float*)(ws + OFF_TOPW);
  int*    tokenOf = (int*)(ws + OFF_TOKEN);
  float*  wOf     = (float*)(ws + OFF_WOF);
  __bf16* Axh  = (__bf16*)(ws + OFF_AXH);
  __bf16* Axl  = (__bf16*)(ws + OFF_AXL);
  __bf16* actH = (__bf16*)(ws + OFF_ACTH);
  __bf16* actL = (__bf16*)(ws + OFF_ACTL);
  __bf16* W1h  = (__bf16*)(ws + OFF_W1H);
  __bf16* W1l  = (__bf16*)(ws + OFF_W1L);
  __bf16* W3h  = (__bf16*)(ws + OFF_W3H);
  __bf16* W3l  = (__bf16*)(ws + OFF_W3L);
  __bf16* W2h  = (__bf16*)(ws + OFF_W2H);
  __bf16* W2l  = (__bf16*)(ws + OFF_W2L);

  moe_zero_kernel<<<(T_ * H_ + 255) / 256, 256, 0, stream>>>(out, counts, tokenOf);
  moe_router_kernel<<<T_ / 8, 256, 0, stream>>>(x, gw, counts, topIdx, topW);
  moe_scan_kernel<<<1, 32, 0, stream>>>(counts, cursor, offsets);
  moe_fill_kernel<<<(T_ + 255) / 256, 256, 0, stream>>>(topIdx, topW, offsets, cursor,
                                                        tokenOf, wOf);
  moe_swizzle_w_kernel<<<dim3(KT_H * 2, E_ * N16_F), 256, 0, stream>>>(w1, W1h, W1l,
                                                                      N16_F, KT_H, H_);
  moe_swizzle_w_kernel<<<dim3(KT_H * 2, E_ * N16_F), 256, 0, stream>>>(w3, W3h, W3l,
                                                                      N16_F, KT_H, H_);
  moe_swizzle_w_kernel<<<dim3(KT_F * 2, E_ * N16_H), 256, 0, stream>>>(w2, W2h, W2l,
                                                                      N16_H, KT_F, F_);
  moe_swizzle_x_kernel<<<dim3(KT_H * 2, PR16), 256, 0, stream>>>(x, tokenOf, Axh, Axl);

  dim3 g1(F_ / BN, T_ / BM, E_);
  moe_up_kernel<<<g1, 256, 0, stream>>>(Axh, Axl, W1h, W1l, W3h, W3l, counts, offsets,
                                        actH, actL);
  dim3 g2(H_ / BN, T_ / BM, E_);
  moe_down_kernel<<<g2, 256, 0, stream>>>(actH, actL, W2h, W2l, counts, offsets,
                                          tokenOf, wOf, out);
}
